// MOEOrthAwareMissingEmbeddingGenerator_29764123361486
// MI455X (gfx1250) — matrix operand order repaired and checked by reading, not compiled, not run on hardware
//
#include <hip/hip_runtime.h>
#include <hip/hip_bf16.h>
#include <math.h>

#define B_  512
#define S_  23
#define D_  768
#define E_  8
#define NH_ 8
#define HD_ 32
#define R_  256
#define L_  50
#define T_  (B_ * S_)      // 11776
#define DH_ (D_ / 2)       // 384
#define LN_EPS 1e-5f

typedef __attribute__((ext_vector_type(16))) _Float16 v16h;
typedef __attribute__((ext_vector_type(8)))  float    v8f;

// ---------------- block reduction (blockDim == 256) ----------------
__device__ __forceinline__ float block_sum(float v, float* buf) {
  int tid = threadIdx.x;
  buf[tid] = v;
  __syncthreads();
  for (int s = 128; s > 0; s >>= 1) {
    if (tid < s) buf[tid] += buf[tid + s];
    __syncthreads();
  }
  float r = buf[0];
  __syncthreads();
  return r;
}

// ---------------- WMMA fragment loaders (CDNA5 16-bit layouts) ----------------
// A 16x32 f16: lanes 0-15 -> M=lane, halves j: K = j + (j>=8 ? 8:0); lanes 16-31: +8
__device__ __forceinline__ v16h load_a16x32(const float* __restrict__ A, int lda,
                                            int m0, int k0, int lane) {
  const float* row = A + (size_t)(m0 + (lane & 15)) * lda + k0 + 8 * (lane >> 4);
  v16h a;
#pragma unroll
  for (int j = 0; j < 16; j += 2) {
    int k = j + ((j >= 8) ? 8 : 0);
    a[j]     = (_Float16)row[k];
    a[j + 1] = (_Float16)row[k + 1];
  }
  return a;
}
// B 32x16 f16: lanes 0-15 -> N=lane, halves j: K = j + (j>=8 ? 8:0); lanes 16-31: +8
__device__ __forceinline__ v16h load_b32x16(const float* __restrict__ Bm, int ldb,
                                            int k0, int n0, int lane) {
  const float* col = Bm + (size_t)(k0 + 8 * (lane >> 4)) * ldb + n0 + (lane & 15);
  v16h b;
#pragma unroll
  for (int j = 0; j < 16; ++j) b[j] = (_Float16)col[(size_t)(j + ((j >= 8) ? 8 : 0)) * ldb];
  return b;
}

__device__ __forceinline__ v8f wmma_f16(v16h a, v16h b, v8f c) {
  return __builtin_amdgcn_wmma_f32_16x16x32_f16(false, a, false, b, (short)0, c,
                                                false, false);
}

// ---------------- generic WMMA GEMM: C[M,N] = A[M,K] * B[K,N] (+bias)(+GELU) ----
// one wave (32 threads) per 32x32 C block: 2x2 register tiling -> each A/B
// fragment feeds 2 WMMAs, 4 back-to-back v_wmma per K-step. blockIdx.z batches.
__global__ void wmma_gemm(const float* __restrict__ A, const float* __restrict__ Bm,
                          float* __restrict__ C, const float* __restrict__ bias,
                          int M, int N, int K, int gelu,
                          long sA, long sB, long sC) {
  const float* Ab = A  + (long)blockIdx.z * sA;
  const float* Bb = Bm + (long)blockIdx.z * sB;
  float*       Cb = C  + (long)blockIdx.z * sC;
  int n0 = blockIdx.x * 32;
  int m0 = blockIdx.y * 32;
  int lane = threadIdx.x;
  v8f acc00 = {}, acc01 = {}, acc10 = {}, acc11 = {};
  for (int k0 = 0; k0 < K; k0 += 32) {
    if (k0 + 32 < K) {
      __builtin_prefetch(Ab + (size_t)(m0 + (lane & 15)) * K + k0 + 32, 0, 3);
      __builtin_prefetch(Ab + (size_t)(m0 + 16 + (lane & 15)) * K + k0 + 32, 0, 3);
    }
    v16h a0 = load_a16x32(Ab, K, m0,      k0, lane);
    v16h a1 = load_a16x32(Ab, K, m0 + 16, k0, lane);
    v16h b0 = load_b32x16(Bb, N, k0, n0,      lane);
    v16h b1 = load_b32x16(Bb, N, k0, n0 + 16, lane);
    acc00 = wmma_f16(a0, b0, acc00);
    acc01 = wmma_f16(a0, b1, acc01);
    acc10 = wmma_f16(a1, b0, acc10);
    acc11 = wmma_f16(a1, b1, acc11);
  }
  // C layout per 16x16 tile: lane l, vgpr i -> m = i + 8*(l>>4), n = l&15
  v8f accs[4] = {acc00, acc01, acc10, acc11};
#pragma unroll
  for (int mi = 0; mi < 2; ++mi) {
#pragma unroll
    for (int ni = 0; ni < 2; ++ni) {
      v8f acc = accs[mi * 2 + ni];
      int n  = n0 + 16 * ni + (lane & 15);
      int mb = m0 + 16 * mi + 8 * (lane >> 4);
      float bv = bias ? bias[n] : 0.0f;
#pragma unroll
      for (int i = 0; i < 8; ++i) {
        float x = acc[i] + bv;
        if (gelu) x = 0.5f * x * (1.0f + erff(x * 0.70710678118654752f));
        Cb[(size_t)(mb + i) * N + n] = x;
      }
    }
  }
}

// ---------------- masked mean context: ctx[B,D] ----------------
__global__ void context_kernel(const float* __restrict__ emb, const int* __restrict__ ex,
                               float* __restrict__ ctx) {
  int b = blockIdx.x, tid = threadIdx.x;
  float cnt = 0.f;
  for (int s = 0; s < S_; ++s) cnt += (float)ex[b * S_ + s];
  float inv = 1.f / (cnt + 1e-8f);
#pragma unroll
  for (int j = 0; j < 3; ++j) {
    int d = tid + 256 * j;
    float a = 0.f;
    for (int s = 0; s < S_; ++s)
      if (ex[b * S_ + s]) a += emb[((size_t)(b * S_ + s)) * D_ + d];
    ctx[(size_t)b * D_ + d] = a * inv;
  }
}

// ---------------- DPE: pos[S,E,D] (default depends only on s!) ----------------
__global__ void dpe_kernel(const float* __restrict__ mt,
                           const float* __restrict__ w0, const float* __restrict__ b0,
                           const float* __restrict__ w1, const float* __restrict__ b1,
                           const float* __restrict__ w2, const float* __restrict__ b2,
                           const float* __restrict__ wo, const float* __restrict__ bo,
                           const float* __restrict__ lng, const float* __restrict__ lnb,
                           float* __restrict__ pos) {
  __shared__ float sdef[D_];
  __shared__ float hA[R_];
  __shared__ float hB[R_];
  __shared__ float red[256];
  int se = blockIdx.x, s = se / E_, e = se % E_, tid = threadIdx.x;
#pragma unroll
  for (int j = 0; j < 3; ++j) sdef[tid + 256 * j] = mt[(size_t)s * D_ + tid + 256 * j];
  __syncthreads();
  {
    const float* w = w0 + (size_t)e * D_ * R_;
    float a = b0[e * R_ + tid];
    for (int d = 0; d < D_; ++d) a += sdef[d] * w[(size_t)d * R_ + tid];
    hA[tid] = fmaxf(a, 0.f);
  }
  __syncthreads();
  {
    const float* w = w1 + (size_t)e * R_ * R_;
    float a = b1[e * R_ + tid];
    for (int k = 0; k < R_; ++k) a += hA[k] * w[k * R_ + tid];
    hB[tid] = fmaxf(a, 0.f);
  }
  __syncthreads();
  {
    const float* w = w2 + (size_t)e * R_ * R_;
    float a = b2[e * R_ + tid];
    for (int k = 0; k < R_; ++k) a += hB[k] * w[k * R_ + tid];
    hA[tid] = fmaxf(a, 0.f);
  }
  __syncthreads();
  float val[3];
  const float* w = wo + (size_t)e * R_ * D_;
#pragma unroll
  for (int j = 0; j < 3; ++j) {
    int d = tid + 256 * j;
    float a = bo[e * D_ + d] + sdef[d];
    for (int k = 0; k < R_; ++k) a += hA[k] * w[(size_t)k * D_ + d];
    val[j] = a;
  }
  float mean = block_sum(val[0] + val[1] + val[2], red) * (1.f / D_);
  float vs = 0.f;
#pragma unroll
  for (int j = 0; j < 3; ++j) vs += (val[j] - mean) * (val[j] - mean);
  float var = block_sum(vs, red) * (1.f / D_);
  float rstd = rsqrtf(var + LN_EPS);
#pragma unroll
  for (int j = 0; j < 3; ++j) {
    int d = tid + 256 * j;
    pos[(size_t)se * D_ + d] = (val[j] - mean) * rstd * lng[e * D_ + d] + lnb[e * D_ + d];
  }
}

// ---------------- s,e-dependent halves of fusion/gate layer 1 ----------------
__global__ void sepos_kernel(const float* __restrict__ pos,
                             const float* __restrict__ fw1p, const float* __restrict__ gw1p,
                             float* __restrict__ vfus, float* __restrict__ vgate) {
  int se = blockIdx.x, e = se % E_, tid = threadIdx.x;
  const float* p  = pos  + (size_t)se * D_;
  const float* wf = fw1p + (size_t)e * D_ * R_;
  const float* wg = gw1p + (size_t)e * D_ * R_;
  float af = 0.f, ag = 0.f;
  for (int d = 0; d < D_; ++d) {
    float pv = p[d];
    af += pv * wf[(size_t)d * R_ + tid];
    ag += pv * wg[(size_t)d * R_ + tid];
  }
  vfus[(size_t)se * R_ + tid]  = af;
  vgate[(size_t)se * R_ + tid] = ag;
}

// ---------------- router default-half: ar[S,R] ----------------
__global__ void ar_kernel(const float* __restrict__ mt, const float* __restrict__ rw1,
                          float* __restrict__ ar) {
  int s = blockIdx.x, tid = threadIdx.x;
  float a = 0.f;
  for (int d = 0; d < D_; ++d) a += mt[(size_t)s * D_ + d] * rw1[(size_t)d * R_ + tid];
  ar[s * R_ + tid] = a;
}

// ---------------- router: softmax + top-2 per token ----------------
__global__ void router_kernel(const float* __restrict__ ar, const float* __restrict__ br,
                              const float* __restrict__ rb1, const float* __restrict__ rw2,
                              const float* __restrict__ rb2,
                              float* __restrict__ topw, int* __restrict__ topi) {
  int t = blockIdx.x, b = t / S_, s = t % S_, tid = threadIdx.x;
  __shared__ float sacc[E_];
  if (tid < E_) sacc[tid] = 0.f;
  __syncthreads();
  float h = fmaxf(ar[s * R_ + tid] + br[b * R_ + tid] + rb1[tid], 0.f);
#pragma unroll
  for (int e = 0; e < E_; ++e) atomicAdd(&sacc[e], h * rw2[tid * E_ + e]);
  __syncthreads();
  if (tid == 0) {
    float sc[E_], mx = -1e30f;
    for (int e = 0; e < E_; ++e) { sc[e] = sacc[e] + rb2[e]; mx = fmaxf(mx, sc[e]); }
    float den = 0.f;
    for (int e = 0; e < E_; ++e) { sc[e] = expf(sc[e] - mx); den += sc[e]; }
    for (int e = 0; e < E_; ++e) sc[e] /= den;
    int i1 = 0;
    for (int e = 1; e < E_; ++e) if (sc[e] > sc[i1]) i1 = e;
    int i2 = -1;
    for (int e = 0; e < E_; ++e) if (e != i1 && (i2 < 0 || sc[e] > sc[i2])) i2 = e;
    topi[t * 2] = i1; topi[t * 2 + 1] = i2;
    topw[t * 2] = sc[i1]; topw[t * 2 + 1] = sc[i2];
  }
}

// ---------------- experts: only missing tokens, only top-2 experts ----------------
__global__ void expert_kernel(const float* __restrict__ emb, const int* __restrict__ ex,
                              const float* __restrict__ ctx,
                              const float* __restrict__ ufus, const float* __restrict__ ugate,
                              const float* __restrict__ vfus, const float* __restrict__ vgate,
                              const float* __restrict__ fb1, const float* __restrict__ fw2,
                              const float* __restrict__ fb2,
                              const float* __restrict__ gb1, const float* __restrict__ gw2,
                              const float* __restrict__ gb2,
                              const float* __restrict__ elng, const float* __restrict__ elnb,
                              const float* __restrict__ topw, const int* __restrict__ topi,
                              float* __restrict__ sec) {
  int t = blockIdx.x, tid = threadIdx.x;
  int b = t / S_, s = t % S_;
  if (ex[t]) {  // present section: pass through (uniform branch per block)
#pragma unroll
    for (int j = 0; j < 3; ++j)
      sec[(size_t)t * D_ + tid + 256 * j] = emb[(size_t)t * D_ + tid + 256 * j];
    return;
  }
  __shared__ float hf[R_];
  __shared__ float hg[R_];
  __shared__ float red[256];
  float out3[3] = {0.f, 0.f, 0.f};
  float cvals[3];
#pragma unroll
  for (int j = 0; j < 3; ++j) cvals[j] = ctx[(size_t)b * D_ + tid + 256 * j];
  for (int ke = 0; ke < 2; ++ke) {
    int e = topi[t * 2 + ke];
    float wgt = topw[t * 2 + ke];
    hf[tid] = fmaxf(ufus[((size_t)e * B_ + b) * R_ + tid] +
                    vfus[((size_t)s * E_ + e) * R_ + tid] + fb1[e * R_ + tid], 0.f);
    hg[tid] = fmaxf(ugate[((size_t)e * B_ + b) * R_ + tid] +
                    vgate[((size_t)s * E_ + e) * R_ + tid] + gb1[e * R_ + tid], 0.f);
    __syncthreads();
    float gsum = block_sum(hg[tid] * gw2[e * R_ + tid], red);
    float gate = 1.f / (1.f + expf(-(gsum + gb2[e])));
    float val[3];
    const float* w2 = fw2 + (size_t)e * R_ * D_;
#pragma unroll
    for (int j = 0; j < 3; ++j) {
      int d = tid + 256 * j;
      float c = fb2[e * D_ + d];
      for (int r = 0; r < R_; ++r) c += hf[r] * w2[(size_t)r * D_ + d];
      val[j] = gate * c + (1.f - gate) * cvals[j];
    }
    float mean = block_sum(val[0] + val[1] + val[2], red) * (1.f / D_);
    float vs = 0.f;
#pragma unroll
    for (int j = 0; j < 3; ++j) vs += (val[j] - mean) * (val[j] - mean);
    float var = block_sum(vs, red) * (1.f / D_);
    float rstd = rsqrtf(var + LN_EPS);
#pragma unroll
    for (int j = 0; j < 3; ++j) {
      int d = tid + 256 * j;
      out3[j] += wgt * ((val[j] - mean) * rstd * elng[e * D_ + d] + elnb[e * D_ + d]);
    }
    __syncthreads();
  }
#pragma unroll
  for (int j = 0; j < 3; ++j) sec[(size_t)t * D_ + tid + 256 * j] = out3[j];
}

// ---------------- LayerNorm over R=256 (reducer output) ----------------
__global__ void ln256_kernel(float* __restrict__ x, const float* __restrict__ g,
                             const float* __restrict__ b) {
  int t = blockIdx.x, tid = threadIdx.x;
  __shared__ float red[256];
  float v = x[(size_t)t * R_ + tid];
  float mean = block_sum(v, red) * (1.f / R_);
  float var = block_sum((v - mean) * (v - mean), red) * (1.f / R_);
  x[(size_t)t * R_ + tid] = (v - mean) * rsqrtf(var + LN_EPS) * g[tid] + b[tid];
}

// ---------------- MHA: one block per (b, head), S=23, hd=32 ----------------
__global__ void attn_kernel(const float* __restrict__ q, const float* __restrict__ k,
                            const float* __restrict__ v, float* __restrict__ o) {
  int bh = blockIdx.x, b = bh / NH_, h = bh % NH_, tid = threadIdx.x;
  __shared__ float sq[S_ * HD_], sk[S_ * HD_], sv[S_ * HD_], sc[S_ * S_];
  for (int i = tid; i < S_ * HD_; i += 256) {
    int s = i / HD_, d = i % HD_;
    size_t idx = ((size_t)(b * S_ + s)) * R_ + h * HD_ + d;
    sq[i] = q[idx]; sk[i] = k[idx]; sv[i] = v[idx];
  }
  __syncthreads();
  const float scale = 0.17677669529663687f;  // 1/sqrt(32)
  for (int i = tid; i < S_ * S_; i += 256) {
    int si = i / S_, sj = i % S_;
    float a = 0.f;
#pragma unroll
    for (int d = 0; d < HD_; ++d) a += sq[si * HD_ + d] * sk[sj * HD_ + d];
    sc[i] = a * scale;
  }
  __syncthreads();
  if (tid < S_) {
    float mx = -1e30f;
    for (int j = 0; j < S_; ++j) mx = fmaxf(mx, sc[tid * S_ + j]);
    float den = 0.f;
    for (int j = 0; j < S_; ++j) { float e = expf(sc[tid * S_ + j] - mx); sc[tid * S_ + j] = e; den += e; }
    float inv = 1.f / den;
    for (int j = 0; j < S_; ++j) sc[tid * S_ + j] *= inv;
  }
  __syncthreads();
  for (int i = tid; i < S_ * HD_; i += 256) {
    int s = i / HD_, d = i % HD_;
    float a = 0.f;
    for (int j = 0; j < S_; ++j) a += sc[s * S_ + j] * sv[j * HD_ + d];
    o[((size_t)(b * S_ + s)) * R_ + h * HD_ + d] = a;
  }
}

// ---------------- mean over S + predictor ----------------
__global__ void meanpred_kernel(const float* __restrict__ op, const float* __restrict__ pw,
                                const float* __restrict__ pb, float* __restrict__ out) {
  int b = blockIdx.x, tid = threadIdx.x;
  __shared__ float om[R_];
  float a = 0.f;
  for (int s = 0; s < S_; ++s) a += op[((size_t)(b * S_ + s)) * R_ + tid];
  om[tid] = a * (1.f / S_);
  __syncthreads();
  if (tid < L_) {
    float acc = pb[tid];
    for (int r = 0; r < R_; ++r) acc += om[r] * pw[r * L_ + tid];
    out[b * L_ + tid] = acc;
  }
}

// ---------------- host-side GEMM launcher (32x32 per wave) ----------------
static inline void gemm(const float* A, const float* Bm, float* C, const float* bias,
                        int M, int N, int K, int gelu, int batch,
                        long sA, long sB, long sC, hipStream_t st) {
  dim3 g(N / 32, M / 32, batch);
  wmma_gemm<<<g, 32, 0, st>>>(A, Bm, C, bias, M, N, K, gelu, sA, sB, sC);
}

extern "C" void kernel_launch(void* const* d_in, const int* in_sizes, int n_in,
                              void* d_out, int out_size, void* d_ws, size_t ws_size,
                              hipStream_t stream) {
  const float* emb      = (const float*)d_in[0];
  const int*   existing = (const int*)  d_in[1];
  const float* mt       = (const float*)d_in[2];
  const float* dpe_w0 = (const float*)d_in[3];  const float* dpe_b0 = (const float*)d_in[4];
  const float* dpe_w1 = (const float*)d_in[5];  const float* dpe_b1 = (const float*)d_in[6];
  const float* dpe_w2 = (const float*)d_in[7];  const float* dpe_b2 = (const float*)d_in[8];
  const float* dpe_wo = (const float*)d_in[9];  const float* dpe_bo = (const float*)d_in[10];
  const float* dpe_lng = (const float*)d_in[11]; const float* dpe_lnb = (const float*)d_in[12];
  const float* fus_w1c = (const float*)d_in[13]; const float* fus_w1p = (const float*)d_in[14];
  const float* fus_b1  = (const float*)d_in[15]; const float* fus_w2  = (const float*)d_in[16];
  const float* fus_b2  = (const float*)d_in[17];
  const float* gate_w1c = (const float*)d_in[18]; const float* gate_w1p = (const float*)d_in[19];
  const float* gate_b1  = (const float*)d_in[20]; const float* gate_w2  = (const float*)d_in[21];
  const float* gate_b2  = (const float*)d_in[22];
  const float* exp_lng = (const float*)d_in[23]; const float* exp_lnb = (const float*)d_in[24];
  const float* rtr_w1 = (const float*)d_in[25]; const float* rtr_b1 = (const float*)d_in[26];
  const float* rtr_w2 = (const float*)d_in[27]; const float* rtr_b2 = (const float*)d_in[28];
  const float* red_w1 = (const float*)d_in[29]; const float* red_b1 = (const float*)d_in[30];
  const float* red_w2 = (const float*)d_in[31]; const float* red_b2 = (const float*)d_in[32];
  const float* red_lng = (const float*)d_in[33]; const float* red_lnb = (const float*)d_in[34];
  const float* qkv_w = (const float*)d_in[35]; const float* qkv_b = (const float*)d_in[36];
  const float* out_w = (const float*)d_in[37]; const float* out_b = (const float*)d_in[38];
  const float* pred_w = (const float*)d_in[39]; const float* pred_b = (const float*)d_in[40];

  // ---- workspace arena (floats), with liveness-based aliasing; ~90 MB total ----
  float* ws = (float*)d_ws;
  size_t off = 0;
  auto alloc = [&](size_t n) { float* p = ws + off; off += n; return p; };
  float* ctx   = alloc((size_t)B_ * D_);
  float* pos   = alloc((size_t)S_ * E_ * D_);
  float* ufus  = alloc((size_t)E_ * B_ * R_);
  float* ugate = alloc((size_t)E_ * B_ * R_);
  float* vfus  = alloc((size_t)S_ * E_ * R_);
  float* vgate = alloc((size_t)S_ * E_ * R_);
  float* ar    = alloc((size_t)S_ * R_);
  float* br    = alloc((size_t)B_ * R_);
  float* topw  = alloc((size_t)T_ * 2);
  int*   topi  = (int*)alloc((size_t)T_ * 2);
  float* sec   = alloc((size_t)T_ * D_);   // D_ == 3*R_: region later holds x | attn-o | proj-o
  float* red1  = alloc((size_t)T_ * DH_);  // reducer hidden; later reused as q
  float* kb    = alloc((size_t)T_ * R_);
  float* vb    = alloc((size_t)T_ * R_);
  float* xbuf  = sec;                      // alias: sec dead after reducer GEMM1
  float* qb    = red1;                     // alias: red1 dead after reducer GEMM2
  float* ob    = sec + (size_t)T_ * R_;
  float* opb   = sec + (size_t)2 * T_ * R_;

  // ---- stage 1: context, DPE (S-only), router halves ----
  context_kernel<<<B_, 256, 0, stream>>>(emb, existing, ctx);
  dpe_kernel<<<S_ * E_, 256, 0, stream>>>(mt, dpe_w0, dpe_b0, dpe_w1, dpe_b1,
                                          dpe_w2, dpe_b2, dpe_wo, dpe_bo,
                                          dpe_lng, dpe_lnb, pos);
  sepos_kernel<<<S_ * E_, 256, 0, stream>>>(pos, fus_w1p, gate_w1p, vfus, vgate);
  ar_kernel<<<S_, 256, 0, stream>>>(mt, rtr_w1, ar);

  // ---- stage 2: batched WMMA GEMMs for ctx-dependent layer-1 halves + router ----
  gemm(ctx, fus_w1c,  ufus,  nullptr, B_, R_, D_, 0, E_, 0, (long)D_ * R_, (long)B_ * R_, stream);
  gemm(ctx, gate_w1c, ugate, nullptr, B_, R_, D_, 0, E_, 0, (long)D_ * R_, (long)B_ * R_, stream);
  gemm(ctx, rtr_w1 + (size_t)D_ * R_, br, nullptr, B_, R_, D_, 0, 1, 0, 0, 0, stream);

  // ---- stage 3: routing + sparse expert evaluation (missing tokens only) ----
  router_kernel<<<T_, 256, 0, stream>>>(ar, br, rtr_b1, rtr_w2, rtr_b2, topw, topi);
  expert_kernel<<<T_, 256, 0, stream>>>(emb, existing, ctx, ufus, ugate, vfus, vgate,
                                        fus_b1, fus_w2, fus_b2, gate_b1, gate_w2, gate_b2,
                                        exp_lng, exp_lnb, topw, topi, sec);

  // ---- stage 4: reducer (WMMA + exact-GELU epilogue) + LN ----
  gemm(sec,  red_w1, red1, red_b1, T_, DH_, D_,  1, 1, 0, 0, 0, stream);
  gemm(red1, red_w2, xbuf, red_b2, T_, R_,  DH_, 1, 1, 0, 0, 0, stream);
  ln256_kernel<<<T_, 256, 0, stream>>>(xbuf, red_lng, red_lnb);

  // ---- stage 5: QKV (WMMA), attention, output projection (WMMA) ----
  gemm(xbuf, qkv_w,                         qb, qkv_b,          T_, R_, R_, 0, 1, 0, 0, 0, stream);
  gemm(xbuf, qkv_w + (size_t)R_ * R_,       kb, qkv_b + R_,     T_, R_, R_, 0, 1, 0, 0, 0, stream);
  gemm(xbuf, qkv_w + (size_t)2 * R_ * R_,   vb, qkv_b + 2 * R_, T_, R_, R_, 0, 1, 0, 0, 0, stream);
  attn_kernel<<<B_ * NH_, 256, 0, stream>>>(qb, kb, vb, ob);
  gemm(ob, out_w, opb, out_b, T_, R_, R_, 0, 1, 0, 0, 0, stream);

  // ---- stage 6: mean over sections + predictor -> d_out [B, L] ----
  meanpred_kernel<<<B_, 256, 0, stream>>>(opb, pred_w, pred_b, (float*)d_out);
}
